// SelfAttention_28518582845514
// MI455X (gfx1250) — matrix operand order repaired and checked by reading, not compiled, not run on hardware
//
#include <hip/hip_runtime.h>
#include <hip/hip_bf16.h>

// Problem constants (from reference): B=16, H=W=64, C=256, CK=C/8=32
#define B_  16
#define N_  4096          // H*W
#define C_  256
#define CK_ 32
#define M_  (B_ * N_)     // 65536 total rows of x

typedef __attribute__((ext_vector_type(16))) __bf16 v16bf;
typedef __attribute__((ext_vector_type(8)))  float  v8f;

union Frag { v16bf v; unsigned u[8]; };

// A-matrix (16x32 bf16) K index for dword d, lane-half h:
//   VGPR 0-3: K = 2d + 8h ; VGPR 4-7: K = 16 + 2(d-4) + 8h   (ISA 7.12.2)
__device__ __forceinline__ int kA(int d, int h) {
  return (d < 4) ? (2 * d + 8 * h) : (16 + 2 * (d - 4) + 8 * h);
}
// B-matrix (32x16 bf16) K index: same K order as the A fragment
//   VGPR 0-3: K = 2d + 8h ; VGPR 4-7: K = 16 + 2(d-4) + 8h
__device__ __forceinline__ int kB(int d, int h) { return kA(d, h); }

__device__ __forceinline__ unsigned short bfbits(float f) {
  union { __bf16 b; unsigned short u; } r; r.b = (__bf16)f; return r.u;
}
__device__ __forceinline__ unsigned pack2(float lo, float hi) {
  union { __bf16 b[2]; unsigned u; } r;
  r.b[0] = (__bf16)lo; r.b[1] = (__bf16)hi;
  return r.u;
}

// xor-shuffle within each 16-lane half. Prefer v_permlane16_b32 (VALU) over
// ds_bpermute so the LDS pipe stays free for async writes + fragment loads.
__device__ __forceinline__ constexpr unsigned sel_lo(int m) {
  unsigned s = 0;
  for (int i = 0; i < 8; ++i) s |= (unsigned)((i ^ m) & 15) << (4 * i);
  return s;
}
__device__ __forceinline__ constexpr unsigned sel_hi(int m) {
  unsigned s = 0;
  for (int i = 8; i < 16; ++i) s |= (unsigned)((i ^ m) & 15) << (4 * (i - 8));
  return s;
}
template <int MSK>
__device__ __forceinline__ float xor16(float v) {
#if __has_builtin(__builtin_amdgcn_permlane16)
  const unsigned u = __builtin_bit_cast(unsigned, v);
  const unsigned r = __builtin_amdgcn_permlane16(u, u, sel_lo(MSK), sel_hi(MSK),
                                                 false, false);
  return __builtin_bit_cast(float, r);
#else
  return __shfl_xor(v, MSK, 32);
#endif
}
__device__ __forceinline__ float rowmax16(float v) {
  v = fmaxf(v, xor16<1>(v)); v = fmaxf(v, xor16<2>(v));
  v = fmaxf(v, xor16<4>(v)); v = fmaxf(v, xor16<8>(v));
  return v;
}
__device__ __forceinline__ float rowsum16(float v) {
  v += xor16<1>(v); v += xor16<2>(v); v += xor16<4>(v); v += xor16<8>(v);
  return v;
}

// gfx1250 async memory->LDS copy (16B per lane), tracked by ASYNCcnt.
__device__ __forceinline__ void async_ld16(const void* g, unsigned ldsOff) {
  asm volatile("global_load_async_to_lds_b128 %0, %1, off"
               :: "v"(ldsOff), "v"((unsigned long long)(size_t)g)
               : "memory");
}
template <int Npend>
__device__ __forceinline__ void wait_async() {
  asm volatile("s_wait_asynccnt %0" :: "i"(Npend) : "memory");
}

// ---------------------------------------------------------------------------
// Projection GEMM: Y[M,NC] = bf16(X[M,C] @ W[C,NC] + bias); TR=true stores
// transposed per batch as Y[B,NC,N] (used for v so the P@V B-fragments read
// contiguous K-pairs from LDS). One 16x16 tile per wave, 8 waves per block.
// ---------------------------------------------------------------------------
template <int NC, bool TR>
__global__ __launch_bounds__(256) void proj_wmma(
    const float* __restrict__ X, const float* __restrict__ W,
    const float* __restrict__ bias, unsigned short* __restrict__ Y,
    int nTiles)
{
  const int lane = threadIdx.x & 31;
  const int wave = threadIdx.x >> 5;
  const int tile = blockIdx.x * 8 + wave;
  if (tile >= nTiles) return;              // whole waves only: EXEC all-1s for WMMA
  constexpr int NTN = NC >> 4;
  const int m0   = (tile / NTN) << 4;
  const int n0   = (tile % NTN) << 4;
  const int half = lane >> 4, lr = lane & 15;

  v8f acc = {};
  const float* xr = X + (size_t)(m0 + lr) * C_;
  for (int k0 = 0; k0 < C_; k0 += 32) {
    Frag a, b;
#pragma unroll
    for (int d = 0; d < 8; ++d) {
      const int ka = k0 + kA(d, half);
      const float2 t = *(const float2*)(xr + ka);          // K pairs contiguous
      a.u[d] = pack2(t.x, t.y);
      const int kb = k0 + kB(d, half);
      const float w0 = W[(size_t)kb       * NC + n0 + lr]; // strided (row-major W)
      const float w1 = W[(size_t)(kb + 1) * NC + n0 + lr];
      b.u[d] = pack2(w0, w1);
    }
    acc = __builtin_amdgcn_wmma_f32_16x16x32_bf16(false, a.v, false, b.v,
                                                  (short)0, acc, false, false);
  }
  const float bv = bias[n0 + lr];
#pragma unroll
  for (int i = 0; i < 8; ++i) {            // C/D: row = i + 8*half, col = lr
    const int row = m0 + i + 8 * half;
    const unsigned short h = bfbits(acc[i] + bv);
    if (TR) {
      const int bb = row >> 12;            // row / N_
      const int q  = row & (N_ - 1);
      Y[((size_t)bb * NC + (n0 + lr)) * N_ + q] = h;       // [B, NC, N]
    } else {
      Y[(size_t)row * NC + (n0 + lr)] = h;
    }
  }
}

// ---------------------------------------------------------------------------
// Flash attention: one wave per (batch, 16-query tile); key chunks of 32.
// Staging uses gfx1250 async loads (memory->LDS) with double buffering, so
// the next chunk's HBM traffic overlaps this chunk's WMMA + softmax.
// 36 async b128 ops per chunk (4 for f, 32 for v^T) -- the largest chunk for
// which full double-buffering is expressible (ASYNCcnt wait max = 63).
// ---------------------------------------------------------------------------
#define APEND 36

__device__ __forceinline__ void stage_async(
    unsigned short* FsB, unsigned short* VsB,
    const unsigned short* __restrict__ F, const unsigned short* __restrict__ VT,
    int b, int p0, int lane)
{
  // f chunk [32 keys][32 ck] = 2KB: 128 x 16B tiles, 4 per lane
#pragma unroll
  for (int j = 0; j < 4; ++j) {
    const int t = j * 32 + lane;
    const int row = t >> 2, seg = t & 3;
    async_ld16(F + ((size_t)b * N_ + p0 + row) * CK_ + seg * 8,
               (unsigned)(size_t)FsB + (unsigned)t * 16u);
  }
  // v^T chunk [256 c][32 keys] = 16KB: 1024 x 16B tiles, 32 per lane
#pragma unroll
  for (int j = 0; j < 32; ++j) {
    const int t = j * 32 + lane;
    const int cr = t >> 2, seg = t & 3;
    async_ld16(VT + ((size_t)b * C_ + cr) * N_ + p0 + seg * 8,
               (unsigned)(size_t)VsB + (unsigned)t * 16u);
  }
}

__global__ __launch_bounds__(32) void attn_wmma(
    const float* __restrict__ X, const unsigned short* __restrict__ G,
    const unsigned short* __restrict__ F, const unsigned short* __restrict__ VT,
    const float* __restrict__ gammap, float* __restrict__ out)
{
  __shared__ __align__(16) unsigned short Fs[2][32 * CK_];  // [key][ck]  2x2 KB
  __shared__ __align__(16) unsigned short Vs[2][C_ * 32];   // [c][key]  2x16 KB
  __shared__ __align__(16) unsigned short Ps[16 * 32];      // [q][key]     1 KB

  const int lane = threadIdx.x & 31;
  const int half = lane >> 4, lr = lane & 15;
  const int b  = blockIdx.x >> 8;          // N_/16 = 256 query tiles per batch
  const int q0 = (blockIdx.x & 255) << 4;

  // Q fragment (A-layout), loaded once: g[b, q0+lr, :32]
  Frag aQ;
  {
    const unsigned short* gq = G + ((size_t)b * N_ + q0 + lr) * CK_;
#pragma unroll
    for (int d = 0; d < 8; ++d)
      aQ.u[d] = *(const unsigned*)(gq + kA(d, half));
  }

  float m[8], l[8];
  v8f o[16];
  {
    v8f zero = {};
#pragma unroll
    for (int t = 0; t < 16; ++t) o[t] = zero;
#pragma unroll
    for (int i = 0; i < 8; ++i) { m[i] = -3.0e38f; l[i] = 0.0f; }
  }

  stage_async(Fs[0], Vs[0], F, VT, b, 0, lane);     // prologue: chunk 0

  for (int p0 = 0; p0 < N_; p0 += 32) {
    const int cur = (p0 >> 5) & 1;
    if (p0 + 32 < N_) {
      stage_async(Fs[1 - cur], Vs[1 - cur], F, VT, b, p0 + 32, lane);
      wait_async<APEND>();                 // chunk `cur` landed; next still in flight
    } else {
      wait_async<0>();
    }
    __syncthreads();

    const unsigned short* FsC = Fs[cur];
    const unsigned short* VsC = Vs[cur];

    // S = Q @ K^T  (two 16x16 key tiles, K-dim = CK = 32)
    v8f s0 = {}, s1 = {};
    {
      Frag bF;
#pragma unroll
      for (int d = 0; d < 8; ++d)
        bF.u[d] = *(const unsigned*)(FsC + lr * CK_ + kB(d, half));
      s0 = __builtin_amdgcn_wmma_f32_16x16x32_bf16(false, aQ.v, false, bF.v,
                                                   (short)0, s0, false, false);
#pragma unroll
      for (int d = 0; d < 8; ++d)
        bF.u[d] = *(const unsigned*)(FsC + (16 + lr) * CK_ + kB(d, half));
      s1 = __builtin_amdgcn_wmma_f32_16x16x32_bf16(false, aQ.v, false, bF.v,
                                                   (short)0, s1, false, false);
    }

    // Online softmax. C-layout: row r = i + 8*half lives in one 16-lane half,
    // so xor masks 1,2,4,8 (v_permlane16) reduce each row over its 16 columns.
    float scale[8];
#pragma unroll
    for (int i = 0; i < 8; ++i) {
      const float mt = rowmax16(fmaxf(s0[i], s1[i]));
      const float mn = fmaxf(m[i], mt);
      scale[i] = __expf(m[i] - mn);
      m[i] = mn;
      const float pa = __expf(s0[i] - mn);
      const float pb = __expf(s1[i] - mn);
      const float rs = rowsum16(pa + pb);
      l[i] = l[i] * scale[i] + rs;
      const int row = i + 8 * half;
      Ps[row * 32 + lr]      = bfbits(pa); // C-layout -> LDS [q][key]
      Ps[row * 32 + 16 + lr] = bfbits(pb);
    }
    __syncthreads();

    // P fragment (A-layout) from LDS, then O = O*scale + P @ V
    Frag aP;
#pragma unroll
    for (int d = 0; d < 8; ++d)
      aP.u[d] = *(const unsigned*)(Ps + lr * 32 + kA(d, half));

#pragma unroll
    for (int ct = 0; ct < 16; ++ct) {
#pragma unroll
      for (int i = 0; i < 8; ++i) o[ct][i] *= scale[i];
      Frag bV;
#pragma unroll
      for (int d = 0; d < 8; ++d)
        bV.u[d] = *(const unsigned*)(VsC + (ct * 16 + lr) * 32 + kB(d, half));
      o[ct] = __builtin_amdgcn_wmma_f32_16x16x32_bf16(false, aP.v, false, bV.v,
                                                      (short)0, o[ct], false, false);
    }
    __syncthreads();
  }

  // Epilogue: y = gamma * (O / l) + x
  const float gamma = *gammap;
#pragma unroll
  for (int ct = 0; ct < 16; ++ct) {
#pragma unroll
    for (int i = 0; i < 8; ++i) {
      const int row = q0 + i + 8 * half;
      const int cc  = ct * 16 + lr;
      const size_t idx = ((size_t)b * N_ + row) * C_ + cc;
      out[idx] = gamma * (o[ct][i] / l[i]) + X[idx];
    }
  }
}

// ---------------------------------------------------------------------------
extern "C" void kernel_launch(void* const* d_in, const int* in_sizes, int n_in,
                              void* d_out, int out_size, void* d_ws, size_t ws_size,
                              hipStream_t stream) {
  (void)in_sizes; (void)n_in; (void)out_size; (void)ws_size;
  const float* x     = (const float*)d_in[0];
  const float* Wf    = (const float*)d_in[1];
  const float* bfp   = (const float*)d_in[2];
  const float* Wg    = (const float*)d_in[3];
  const float* bg    = (const float*)d_in[4];
  const float* Wh    = (const float*)d_in[5];
  const float* bh    = (const float*)d_in[6];
  const float* gamma = (const float*)d_in[7];
  float* out = (float*)d_out;

  // Workspace (bf16 bits): g [M,32] 4MB | f [M,32] 4MB | v^T [B,C,N] 32MB
  unsigned short* gbf  = (unsigned short*)d_ws;
  unsigned short* fbf  = gbf + (size_t)M_ * CK_;
  unsigned short* vtbf = fbf + (size_t)M_ * CK_;

  {
    const int tiles = (M_ / 16) * (CK_ / 16);          // 8192
    proj_wmma<CK_, false><<<tiles / 8, 256, 0, stream>>>(x, Wg, bg, gbf, tiles);
    proj_wmma<CK_, false><<<tiles / 8, 256, 0, stream>>>(x, Wf, bfp, fbf, tiles);
  }
  {
    const int tiles = (M_ / 16) * (C_ / 16);           // 65536
    proj_wmma<C_, true><<<tiles / 8, 256, 0, stream>>>(x, Wh, bh, vtbf, tiles);
  }
  attn_wmma<<<B_ * (N_ / 16), 32, 0, stream>>>(x, gbf, fbf, vtbf, gamma, out);
}
